// ChunkedSIEVEModel_7662221656333
// MI455X (gfx1250) — matrix operand order repaired and checked by reading, not compiled, not run on hardware
//
#include <hip/hip_runtime.h>
#include <hip/hip_bf16.h>
#include <math.h>

// Problem constants (from reference)
#define BCH 2048   // B chunks
#define NV  256    // V variants per chunk
#define NF  64     // F features
#define ND  256    // D embed dim
#define NS  256    // S samples
#define POS_SCALE 1e-6f

typedef __bf16 bf16;
typedef __attribute__((ext_vector_type(16))) bf16  v16bf;
typedef __attribute__((ext_vector_type(8)))  bf16  v8bf;
typedef __attribute__((ext_vector_type(8)))  float v8f;

union AFrag { v16bf v; struct { v8bf lo, hi; } h; };

#define LDS_A_STRIDE 72   // 64 K-elems + pad; 144 B row stride keeps 16-B alignment, dodges bank conflicts

// Branch-free tanh: prefer the gfx1250 hardware V_TANH_F32 trans op; otherwise
// a divergence-free rational via the native exp (v_exp_f32). Never the branchy
// ocml tanhf (it wraps EXEC save/restore around every call, poison next to WMMA).
__device__ __forceinline__ float fast_tanh(float x) {
#if __has_builtin(__builtin_amdgcn_tanhf)
    return __builtin_amdgcn_tanhf(x);
#else
    float e = __expf(2.0f * x);          // v_exp_f32 (native), no branches
    return (e - 1.0f) * __frcp_rn(e + 1.0f);
#endif
}

// ---------------------------------------------------------------------------
// Kernel 0: W_feat [F,D] f32 -> Wt [D,F] bf16  (so B-fragments are contiguous)
// ---------------------------------------------------------------------------
__global__ __launch_bounds__(256)
void convert_wfeat_kernel(const float* __restrict__ W, bf16* __restrict__ Wt) {
    int i = blockIdx.x * 256 + threadIdx.x;   // i over F*D
    if (i < NF * ND) {
        int f = i / ND, d = i % ND;
        Wt[d * NF + f] = (bf16)W[i];
    }
}

// ---------------------------------------------------------------------------
// Kernel 1: fused per-chunk embedding (WMMA bf16 GEMM + tanh + pool + gather)
// grid = BCH blocks, 256 threads (8 waves). Each wave owns 2 N-tiles (32 d's).
// ---------------------------------------------------------------------------
__global__ __launch_bounds__(256)
void sieve_embed_kernel(const float* __restrict__ features,   // [B,V,F] f32
                        const int*   __restrict__ positions,  // [B,V]
                        const int*   __restrict__ gene_ids,   // [B,V]
                        const int*   __restrict__ mask,       // [B,V]
                        const float* __restrict__ b_feat,     // [D]
                        const float* __restrict__ gene_table, // [G,D]
                        const float* __restrict__ w_pos,      // [D]
                        const bf16*  __restrict__ WfT,        // [D,F] bf16
                        float*       __restrict__ emb_out)    // [B,D]
{
    __shared__ bf16  sA[NV * LDS_A_STRIDE];   // ~36.9 KB features tile (bf16)
    __shared__ float s_sum[ND];               // pooled tanh-term accumulator
    __shared__ float s_pos[NV];
    __shared__ float s_maskf[NV];
    __shared__ int   s_gid[NV];

    const int b    = blockIdx.x;
    const int tid  = threadIdx.x;
    const int wave = tid >> 5;
    const int lane = tid & 31;

    // ---- stage: per-variant scalars + f32->bf16 conversion of A into LDS ----
    {
        const int v = tid;                               // 256 threads == NV rows
        s_pos[v]   = (float)positions[b * NV + v] * POS_SCALE;
        s_maskf[v] = mask[b * NV + v] ? 1.0f : 0.0f;
        s_gid[v]   = gene_ids[b * NV + v];
        s_sum[v]   = 0.0f;                               // NV == ND == 256
        const float4* src = (const float4*)(features + (size_t)b * NV * NF + (size_t)v * NF);
        bf16* dst = &sA[v * LDS_A_STRIDE];
        #pragma unroll
        for (int i = 0; i < NF / 4; ++i) {
            float4 f4 = src[i];
            dst[i * 4 + 0] = (bf16)f4.x;
            dst[i * 4 + 1] = (bf16)f4.y;
            dst[i * 4 + 2] = (bf16)f4.z;
            dst[i * 4 + 3] = (bf16)f4.w;
        }
    }
    __syncthreads();

    // ---- B fragments (contiguous K order, same as A): lane = {N=lane&15, Khalf=lane>>4},
    //      each lane holds 16 consecutive K values -> one aligned 32-B load. ----
    const int col   = lane & 15;
    const int khalf = lane >> 4;
    const int nt0   = wave * 2;
    const int d0    = nt0 * 16 + col;
    const int d1    = d0 + 16;

    v16bf b00 = *(const v16bf*)(WfT + d0 * NF +  0 + khalf * 16);  // nt0, k 0..31
    v16bf b01 = *(const v16bf*)(WfT + d0 * NF + 32 + khalf * 16);  // nt0, k 32..63
    v16bf b10 = *(const v16bf*)(WfT + d1 * NF +  0 + khalf * 16);  // nt1, k 0..31
    v16bf b11 = *(const v16bf*)(WfT + d1 * NF + 32 + khalf * 16);  // nt1, k 32..63

    const float bf0 = b_feat[d0], wp0 = w_pos[d0];
    const float bf1 = b_feat[d1], wp1 = w_pos[d1];

    float acc0 = 0.0f, acc1 = 0.0f;
    const int mbase = khalf * 8;   // C layout: VGPR r holds M = r (lanes 0-15) / 8+r (lanes 16-31)

    #pragma unroll 2
    for (int mt = 0; mt < 16; ++mt) {
        // A fragment (contiguous K order): lane row M = lane&15;
        // lanes 0-15 hold K {0..15}+kb*32, lanes 16-31 hold K {16..31}+kb*32 (matches B).
        const bf16* arow = &sA[(mt * 16 + col) * LDS_A_STRIDE];
        AFrag a0, a1;
        a0.h.lo = *(const v8bf*)(arow +  0 + khalf * 16);
        a0.h.hi = *(const v8bf*)(arow +  8 + khalf * 16);
        a1.h.lo = *(const v8bf*)(arow + 32 + khalf * 16);
        a1.h.hi = *(const v8bf*)(arow + 40 + khalf * 16);

        v8f c0 = {}; v8f c1 = {};
        c0 = __builtin_amdgcn_wmma_f32_16x16x32_bf16(false, a0.v, false, b00, (short)0, c0, false, false);
        c0 = __builtin_amdgcn_wmma_f32_16x16x32_bf16(false, a1.v, false, b01, (short)0, c0, false, false);
        c1 = __builtin_amdgcn_wmma_f32_16x16x32_bf16(false, a0.v, false, b10, (short)0, c1, false, false);
        c1 = __builtin_amdgcn_wmma_f32_16x16x32_bf16(false, a1.v, false, b11, (short)0, c1, false, false);

        // epilogue: bias + pos*w_pos, tanh, mask, pool over the tile's 16 V rows
        #pragma unroll
        for (int r = 0; r < 8; ++r) {
            const int   v  = mt * 16 + mbase + r;
            const float p  = s_pos[v];
            const float mf = s_maskf[v];
            acc0 += fast_tanh(fmaf(p, wp0, c0[r] + bf0)) * mf;
            acc1 += fast_tanh(fmaf(p, wp1, c1[r] + bf1)) * mf;
        }
    }
    atomicAdd(&s_sum[d0], acc0);   // lane n and lane n+16 share d -> 2-way ds_add_f32
    atomicAdd(&s_sum[d1], acc1);
    __syncthreads();

    // ---- gene_table gather-sum + masked-mean finalize; thread == d ----
    float accg = 0.0f, denom = 0.0f;
    for (int v = 0; v < NV; ++v) {
        const float mf = s_maskf[v];
        denom += mf;
        if (mf != 0.0f)
            accg += gene_table[(size_t)s_gid[v] * ND + tid];
    }
    denom = fmaxf(denom, 1.0f);
    emb_out[(size_t)b * ND + tid] = (s_sum[tid] + accg) / denom;
}

// ---------------------------------------------------------------------------
// Kernel 2: scores = tanh(emb @ W_att1 + b_att1) @ W_att2 + b_att2
// one wave per chunk; lane owns 2 of the 64 hidden units.
// ---------------------------------------------------------------------------
__global__ __launch_bounds__(256)
void attn_score_kernel(const float* __restrict__ emb,
                       const float* __restrict__ W_att1,  // [D,64]
                       const float* __restrict__ b_att1,  // [64]
                       const float* __restrict__ W_att2,  // [64,1]
                       const float* __restrict__ b_att2,  // [1]
                       float*       __restrict__ scores)  // [B]
{
    const int wave = threadIdx.x >> 5;
    const int lane = threadIdx.x & 31;
    const int b    = blockIdx.x * 8 + wave;
    const int j0   = lane * 2;

    float h0 = b_att1[j0], h1 = b_att1[j0 + 1];
    const float* e = emb + (size_t)b * ND;
    for (int d = 0; d < ND; ++d) {
        const float  ev = e[d];                               // wave-uniform broadcast load
        const float2 w  = *(const float2*)(W_att1 + d * 64 + j0);
        h0 = fmaf(ev, w.x, h0);
        h1 = fmaf(ev, w.y, h1);
    }
    const float2 w2 = *(const float2*)(W_att2 + j0);
    float p = fast_tanh(h0) * w2.x + fast_tanh(h1) * w2.y;
    #pragma unroll
    for (int off = 16; off > 0; off >>= 1)
        p += __shfl_xor(p, off, 32);
    if (lane == 0) scores[b] = p + b_att2[0];
}

// ---------------------------------------------------------------------------
// Kernel 3: per-sample segment softmax over chunks + weighted agg + classifier
// one block per sample; thread == d for the aggregation phase.
// ---------------------------------------------------------------------------
__global__ __launch_bounds__(256)
void segment_agg_kernel(const float* __restrict__ emb,     // [B,D]
                        const float* __restrict__ scores,  // [B]
                        const int*   __restrict__ seg,     // [B]
                        const float* __restrict__ W_cls,   // [D,1]
                        const float* __restrict__ b_cls,   // [1]
                        float*       __restrict__ out)     // [S,1]
{
    __shared__ int   s_seg[BCH];
    __shared__ float s_sc[BCH];
    __shared__ float red[256];

    const int s   = blockIdx.x;
    const int tid = threadIdx.x;

    for (int i = tid; i < BCH; i += 256) {
        s_seg[i] = seg[i];
        s_sc[i]  = scores[i];
    }
    __syncthreads();

    // segment max
    float m = -3.0e38f;
    for (int c = tid; c < BCH; c += 256)
        if (s_seg[c] == s) m = fmaxf(m, s_sc[c]);
    red[tid] = m; __syncthreads();
    for (int o = 128; o > 0; o >>= 1) {
        if (tid < o) red[tid] = fmaxf(red[tid], red[tid + o]);
        __syncthreads();
    }
    const float smax = red[0];
    __syncthreads();

    // segment exp-sum
    float es = 0.0f;
    for (int c = tid; c < BCH; c += 256)
        if (s_seg[c] == s) es += __expf(s_sc[c] - smax);
    red[tid] = es; __syncthreads();
    for (int o = 128; o > 0; o >>= 1) {
        if (tid < o) red[tid] += red[tid + o];
        __syncthreads();
    }
    const float ssum = red[0];
    __syncthreads();

    // weighted aggregate of emb rows (thread owns dimension d = tid)
    float agg = 0.0f;
    for (int c = 0; c < BCH; ++c)
        if (s_seg[c] == s)
            agg = fmaf(emb[(size_t)c * ND + tid], __expf(s_sc[c] - smax), agg);
    agg /= ssum;

    // classifier dot product
    red[tid] = agg * W_cls[tid];
    __syncthreads();
    for (int o = 128; o > 0; o >>= 1) {
        if (tid < o) red[tid] += red[tid + o];
        __syncthreads();
    }
    if (tid == 0) out[s] = red[0] + b_cls[0];
}

// ---------------------------------------------------------------------------
extern "C" void kernel_launch(void* const* d_in, const int* in_sizes, int n_in,
                              void* d_out, int out_size, void* d_ws, size_t ws_size,
                              hipStream_t stream) {
    const float* features   = (const float*)d_in[0];
    const int*   positions  = (const int*)  d_in[1];
    const int*   gene_ids   = (const int*)  d_in[2];
    const int*   mask       = (const int*)  d_in[3];
    const int*   seg        = (const int*)  d_in[4];
    const float* W_feat     = (const float*)d_in[5];
    const float* b_feat     = (const float*)d_in[6];
    const float* gene_table = (const float*)d_in[7];
    const float* w_pos      = (const float*)d_in[8];
    const float* W_att1     = (const float*)d_in[9];
    const float* b_att1     = (const float*)d_in[10];
    const float* W_att2     = (const float*)d_in[11];
    const float* b_att2     = (const float*)d_in[12];
    const float* W_cls      = (const float*)d_in[13];
    const float* b_cls      = (const float*)d_in[14];
    float* out = (float*)d_out;

    // workspace layout: emb [B,D] f32 (2 MB) | scores [B] f32 (8 KB) | Wt bf16 (32 KB)
    char*  ws     = (char*)d_ws;
    float* emb    = (float*)ws;
    float* scores = (float*)(ws + (size_t)BCH * ND * sizeof(float));
    bf16*  WfT    = (bf16*) (ws + (size_t)BCH * ND * sizeof(float) + (size_t)BCH * sizeof(float));

    convert_wfeat_kernel<<<(NF * ND + 255) / 256, 256, 0, stream>>>(W_feat, WfT);
    sieve_embed_kernel<<<BCH, 256, 0, stream>>>(features, positions, gene_ids, mask,
                                                b_feat, gene_table, w_pos, WfT, emb);
    attn_score_kernel<<<BCH / 8, 256, 0, stream>>>(emb, W_att1, b_att1, W_att2, b_att2, scores);
    segment_agg_kernel<<<NS, 256, 0, stream>>>(emb, scores, seg, W_cls, b_cls, out);
}
